// CausalSelfAttentionRoPE_61993557950519
// MI455X (gfx1250) — compile-verified, matrix operand order checked by reading, not run on hardware
//
#include <hip/hip_runtime.h>
#include <hip/hip_bf16.h>

#define N_HEADS 16
#define DMODEL  2048
#define DK      128
#define SEQ     2048
#define BATCH   2

typedef __attribute__((ext_vector_type(16))) __bf16 v16bf;
typedef __attribute__((ext_vector_type(8)))  float  v8f;

union Frag { v16bf f; unsigned u[8]; };

// Scheduling-group directives (no-op if unsupported). 0x100 = DS read, 0x8 = MFMA/WMMA.
#if defined(__has_builtin)
#if __has_builtin(__builtin_amdgcn_sched_group_barrier)
#define SGB(mask, size) __builtin_amdgcn_sched_group_barrier((mask), (size), 0)
#endif
#endif
#ifndef SGB
#define SGB(mask, size)
#endif

__device__ __forceinline__ unsigned short f32_bf16(float x) {
    unsigned u = __float_as_uint(x);
    u += 0x7FFFu + ((u >> 16) & 1u);   // round-to-nearest-even
    return (unsigned short)(u >> 16);
}
__device__ __forceinline__ unsigned pack_bf16(float lo, float hi) {
    return (unsigned)f32_bf16(lo) | ((unsigned)f32_bf16(hi) << 16);
}
__device__ __forceinline__ float bf16_f32(unsigned short h) {
    return __uint_as_float(((unsigned)h) << 16);
}

// Load a 16-bit A/B operand fragment from an LDS row (direct array indexing so
// address-space inference keeps these as ds_load_b128 pairs).
#define LDFRAG(fr, ROWEXPR, OFF)                                                  \
    do {                                                                          \
        _Pragma("unroll")                                                         \
        for (int _j = 0; _j < 4; ++_j) {                                          \
            (fr).u[_j]     = *(const unsigned*)&ROWEXPR[(OFF) + kbase + 2 * _j];       \
            (fr).u[4 + _j] = *(const unsigned*)&ROWEXPR[(OFF) + 16 + kbase + 2 * _j];  \
        }                                                                         \
    } while (0)

// ---------------------------------------------------------------- convert
__global__ __launch_bounds__(256)
void convert_f32_bf16(const float* __restrict__ in, unsigned short* __restrict__ out, int n) {
    int i = (blockIdx.x * blockDim.x + threadIdx.x) * 4;
    if (i + 3 < n) {
        float4 v = *(const float4*)(in + i);
        uint2 p;
        p.x = pack_bf16(v.x, v.y);
        p.y = pack_bf16(v.z, v.w);
        *(uint2*)(out + i) = p;
    }
}

// ---------------------------------------------------------------- GEMM: C[M,N] = A[M,K](bf16) * W[N,K]^T (f32 weights -> bf16)
// 256 threads = 8 waves; block tile 128x128; K-step 32; double-buffered LDS.
// Wave w -> 32x64 sub-tile: rows (w&3)*32, cols (w>>2)*64.
template<bool OUT_BF16>
__global__ __launch_bounds__(256)
void gemm_xwT(const unsigned short* __restrict__ A,
              const float* __restrict__ W,
              void* __restrict__ Cout,
              int M, int N, int K) {
    __shared__ unsigned short Asm[2][128][32];
    __shared__ unsigned short Bsm[2][128][32];

    const int tid   = threadIdx.x;
    const int lane  = tid & 31;
    const int wave  = tid >> 5;
    const int rw    = wave & 3;          // row strip (32 rows)
    const int cw    = wave >> 2;         // col strip (64 cols)
    const int lhalf = lane >> 4;
    const int l16   = lane & 15;
    const int kbase = lhalf * 8;         // ISA 16-bit operand K split

    const int m0 = blockIdx.y * 128;
    const int n0 = blockIdx.x * 128;

    const int sr = tid >> 1;             // staging row 0..127
    const int sa = (tid & 1) * 8;        // A: dword offset (16 dwords/row)
    const int sb = (tid & 1) * 16;       // B: f32 element offset

    v8f zf = {};
    v8f acc[2][4];
#pragma unroll
    for (int i = 0; i < 2; ++i)
#pragma unroll
        for (int j = 0; j < 4; ++j) acc[i][j] = zf;

    // ---- prologue: stage k0 = 0 into buffer 0
    {
        const unsigned* srcA = (const unsigned*)(A + (size_t)(m0 + sr) * K);
        unsigned* dstA = (unsigned*)&Asm[0][sr][0];
#pragma unroll
        for (int j = 0; j < 8; ++j) dstA[sa + j] = srcA[sa + j];
        const float* srcW = W + (size_t)(n0 + sr) * K + sb;
        unsigned* dstB = (unsigned*)&Bsm[0][sr][sb];
#pragma unroll
        for (int j = 0; j < 8; ++j) dstB[j] = pack_bf16(srcW[2 * j], srcW[2 * j + 1]);
    }
    __syncthreads();

    for (int k0 = 0; k0 < K; k0 += 32) {
        const int cur = (k0 >> 5) & 1;
        // ---- stage next tile into the other buffer (overlaps with compute below)
        if (k0 + 32 < K) {
            const int nxt = cur ^ 1;
            const unsigned* srcA = (const unsigned*)(A + (size_t)(m0 + sr) * K + k0 + 32);
            unsigned* dstA = (unsigned*)&Asm[nxt][sr][0];
#pragma unroll
            for (int j = 0; j < 8; ++j) dstA[sa + j] = srcA[sa + j];
            const float* srcW = W + (size_t)(n0 + sr) * K + k0 + 32 + sb;
            unsigned* dstB = (unsigned*)&Bsm[nxt][sr][sb];
#pragma unroll
            for (int j = 0; j < 8; ++j) dstB[j] = pack_bf16(srcW[2 * j], srcW[2 * j + 1]);
            if (k0 + 64 < K) {   // warm L2/WGP$ one more step ahead
                __builtin_prefetch(A + (size_t)(m0 + sr) * K + k0 + 64, 0, 0);
                __builtin_prefetch(W + (size_t)(n0 + sr) * K + k0 + 64 + sb, 0, 0);
            }
        }

        // ---- compute: af[2] preloaded, B-fragments ping-ponged (2-deep pipeline)
        Frag af[2];
#pragma unroll
        for (int i = 0; i < 2; ++i)
            LDFRAG(af[i], Asm[cur][rw * 32 + i * 16 + l16], 0);

        Frag bfa, bfb;
        LDFRAG(bfa, Bsm[cur][cw * 64 + l16], 0);
#pragma unroll
        for (int j = 0; j < 4; ++j) {
            Frag& curB = (j & 1) ? bfb : bfa;
            Frag& nxtB = (j & 1) ? bfa : bfb;
            if (j + 1 < 4)
                LDFRAG(nxtB, Bsm[cur][cw * 64 + (j + 1) * 16 + l16], 0);
            acc[0][j] = __builtin_amdgcn_wmma_f32_16x16x32_bf16(
                false, af[0].f, false, curB.f, (short)0, acc[0][j], false, false);
            acc[1][j] = __builtin_amdgcn_wmma_f32_16x16x32_bf16(
                false, af[1].f, false, curB.f, (short)0, acc[1][j], false, false);
        }
        // schedule: af(4)+bf0(2) loads up front, then [2 WMMA | 2 DS-read] pipeline
        SGB(0x100, 6);
#pragma unroll
        for (int j = 0; j < 3; ++j) { SGB(0x008, 2); SGB(0x100, 2); }
        SGB(0x008, 2);

        __syncthreads();
    }

#pragma unroll
    for (int i = 0; i < 2; ++i)
#pragma unroll
        for (int j = 0; j < 4; ++j)
#pragma unroll
            for (int r = 0; r < 8; ++r) {
                int grow = m0 + rw * 32 + i * 16 + r + 8 * lhalf;
                int gcol = n0 + cw * 64 + j * 16 + l16;
                if (OUT_BF16)
                    ((unsigned short*)Cout)[(size_t)grow * N + gcol] = f32_bf16(acc[i][j][r]);
                else
                    ((float*)Cout)[(size_t)grow * N + gcol] = acc[i][j][r];
            }
}

// ---------------------------------------------------------------- RoPE (interleaved pairs), in place on bf16
__global__ __launch_bounds__(256)
void rope_inplace(unsigned short* __restrict__ Q, int BS) {
    int idx = blockIdx.x * blockDim.x + threadIdx.x;
    int total = BS * (DMODEL / 2);
    if (idx >= total) return;
    int row  = idx / (DMODEL / 2);
    int pcol = idx % (DMODEL / 2);     // head h = pcol/64, pair p = pcol%64
    int p = pcol & 63;
    int s = row % SEQ;
    float theta = __powf(10000.0f, -(float)p / 64.0f);  // base^(-2p/d_k)
    float ang = (float)s * theta;
    float cs, sn;
    __sincosf(ang, &sn, &cs);
    unsigned* q = (unsigned*)(Q + (size_t)row * DMODEL + pcol * 2);
    unsigned pr = *q;
    float xe = bf16_f32((unsigned short)(pr & 0xFFFF));
    float xo = bf16_f32((unsigned short)(pr >> 16));
    *q = pack_bf16(xe * cs - xo * sn, xe * sn + xo * cs);
}

// ---------------------------------------------------------------- flash attention (causal, online softmax)
// grid: (S/64, B*H); 128 threads = 4 waves, each wave owns 16 query rows.
__global__ __launch_bounds__(128)
void flash_attn(const unsigned short* __restrict__ Q,
                const unsigned short* __restrict__ Kb,
                const unsigned short* __restrict__ Vb,
                unsigned short* __restrict__ O) {
    __shared__ unsigned short Ksm[64][128];     // [key][d]   -> B operand for QK^T
    __shared__ unsigned short Vsm[128][64];     // [d][key]   -> B operand for P*V
    __shared__ unsigned short Psm[4][16][64];   // per-wave P reshape buffer

    const int tid   = threadIdx.x;
    const int lane  = tid & 31;
    const int wave  = tid >> 5;     // 0..3
    const int lhalf = lane >> 4;
    const int l16   = lane & 15;
    const int kbase = lhalf * 8;

    const int qt = blockIdx.x;
    const int bh = blockIdx.y;
    const int b  = bh / N_HEADS;
    const int h  = bh % N_HEADS;
    const int q0 = qt * 64;

    // Q fragments for this wave's 16 rows (A operand, 4 chunks over d_k=128)
    Frag qf[4];
    {
        const unsigned short* qrow =
            Q + ((size_t)(b * SEQ + q0 + wave * 16 + l16)) * DMODEL + h * DK;
#pragma unroll
        for (int c = 0; c < 4; ++c)
#pragma unroll
            for (int j = 0; j < 4; ++j) {
                qf[c].u[j]     = *(const unsigned*)(qrow + 32 * c + kbase + 2 * j);
                qf[c].u[4 + j] = *(const unsigned*)(qrow + 32 * c + 16 + kbase + 2 * j);
            }
    }

    v8f zf = {};
    v8f accO[8];
#pragma unroll
    for (int t = 0; t < 8; ++t) accO[t] = zf;
    float mrow[8], lrow[8];
#pragma unroll
    for (int r = 0; r < 8; ++r) { mrow[r] = -1e30f; lrow[r] = 0.0f; }

    const float scale = 0.08838834764831845f;   // 1/sqrt(d_k)

    for (int kt = 0; kt <= qt; ++kt) {          // causal: skip upper tiles entirely
        const int kc0 = kt * 64;
        __syncthreads();
        {   // stage K chunk (row-major dword copy; compiler widens to b128)
            int r  = tid >> 1;
            int cd = (tid & 1) * 32;
            const unsigned* src =
                (const unsigned*)(Kb + ((size_t)(b * SEQ + kc0 + r)) * DMODEL + h * DK) + cd;
            unsigned* dst = (unsigned*)&Ksm[r][0] + cd;
#pragma unroll
            for (int j = 0; j < 32; ++j) dst[j] = src[j];
        }
        {   // stage V transposed: thread owns key pair (2kp,2kp+1) x 32 d values; packed b32 stores
            int kp = tid & 31;              // key pair 0..31
            int d0 = (tid >> 5) * 32;       // d chunk per wave
            const unsigned* s0 =
                (const unsigned*)(Vb + ((size_t)(b * SEQ + kc0 + 2 * kp)) * DMODEL + h * DK + d0);
            const unsigned* s1 =
                (const unsigned*)(Vb + ((size_t)(b * SEQ + kc0 + 2 * kp + 1)) * DMODEL + h * DK + d0);
            unsigned ra[16], rb[16];
#pragma unroll
            for (int j = 0; j < 16; ++j) { ra[j] = s0[j]; rb[j] = s1[j]; }
#pragma unroll
            for (int j = 0; j < 16; ++j) {
                *(unsigned*)&Vsm[d0 + 2 * j][2 * kp] =
                    (ra[j] & 0xFFFFu) | (rb[j] << 16);
                *(unsigned*)&Vsm[d0 + 2 * j + 1][2 * kp] =
                    (ra[j] >> 16) | (rb[j] & 0xFFFF0000u);
            }
        }
        __syncthreads();

        // ---- scores S = Q K^T : 16 load->WMMA steps, 2-deep ping-pong pipeline
        v8f sc[4];
#pragma unroll
        for (int n = 0; n < 4; ++n) sc[n] = zf;
        {
            Frag kfa, kfb;
            LDFRAG(kfa, Ksm[l16], 0);                     // c=0, n=0
#pragma unroll
            for (int idx = 0; idx < 16; ++idx) {
                const int c = idx >> 2, n = idx & 3;
                Frag& curF = (idx & 1) ? kfb : kfa;
                Frag& nxtF = (idx & 1) ? kfa : kfb;
                if (idx + 1 < 16) {
                    const int c2 = (idx + 1) >> 2, n2 = (idx + 1) & 3;
                    LDFRAG(nxtF, Ksm[n2 * 16 + l16], 32 * c2);
                }
                sc[n] = __builtin_amdgcn_wmma_f32_16x16x32_bf16(
                    false, qf[c].f, false, curF.f, (short)0, sc[n], false, false);
            }
            // schedule: 2 fragments (4 loads) in flight, then [WMMA | 2 DS-read] pipeline
            SGB(0x100, 4);
#pragma unroll
            for (int i = 0; i < 14; ++i) { SGB(0x008, 1); SGB(0x100, 2); }
            SGB(0x008, 2);
        }

        // ---- online softmax (row values live in a 16-lane half of the C/D layout)
#pragma unroll
        for (int r = 0; r < 8; ++r) {
            int qrow = q0 + wave * 16 + r + 8 * lhalf;
            float sv[4];
            float mx = -1e30f;
#pragma unroll
            for (int n = 0; n < 4; ++n) {
                int kidx = kc0 + n * 16 + l16;
                float v = sc[n][r] * scale;
                if (kidx > qrow) v = -1e30f;
                sv[n] = v;
                mx = fmaxf(mx, v);
            }
#pragma unroll
            for (int off = 1; off < 16; off <<= 1)
                mx = fmaxf(mx, __shfl_xor(mx, off, 32));
            float mnew = fmaxf(mrow[r], mx);
            float corr = __expf(mrow[r] - mnew);
            mrow[r] = mnew;
            float rs = 0.0f;
#pragma unroll
            for (int n = 0; n < 4; ++n) {
                float p = __expf(sv[n] - mnew);
                rs += p;
                Psm[wave][r + 8 * lhalf][n * 16 + l16] = f32_bf16(p);
            }
#pragma unroll
            for (int off = 1; off < 16; off <<= 1)
                rs += __shfl_xor(rs, off, 32);
            lrow[r] = lrow[r] * corr + rs;
#pragma unroll
            for (int t = 0; t < 8; ++t) accO[t][r] *= corr;
        }
        __syncthreads();    // make P visible for A-operand reload

        // ---- O += P @ V : 16 load->WMMA steps, 2-deep ping-pong pipeline
        Frag pf[2];
#pragma unroll
        for (int c = 0; c < 2; ++c)
            LDFRAG(pf[c], Psm[wave][l16], 32 * c);
        {
            Frag vfa, vfb;
            LDFRAG(vfa, Vsm[l16], 0);                     // c=0, t=0
#pragma unroll
            for (int idx = 0; idx < 16; ++idx) {
                const int c = idx >> 3, t = idx & 7;
                Frag& curF = (idx & 1) ? vfb : vfa;
                Frag& nxtF = (idx & 1) ? vfa : vfb;
                if (idx + 1 < 16) {
                    const int c2 = (idx + 1) >> 3, t2 = (idx + 1) & 7;
                    LDFRAG(nxtF, Vsm[t2 * 16 + l16], 32 * c2);
                }
                accO[t] = __builtin_amdgcn_wmma_f32_16x16x32_bf16(
                    false, pf[c].f, false, curF.f, (short)0, accO[t], false, false);
            }
            // schedule: pf(4)+vf0(2)+vf1(2) loads up front, then [WMMA | 2 DS-read]
            SGB(0x100, 8);
#pragma unroll
            for (int i = 0; i < 14; ++i) { SGB(0x008, 1); SGB(0x100, 2); }
            SGB(0x008, 2);
        }
    }

    // ---- normalize + store bf16 [B,S,D] (heads concatenated)
#pragma unroll
    for (int r = 0; r < 8; ++r) {
        float inv = 1.0f / lrow[r];
        int s = q0 + wave * 16 + r + 8 * lhalf;
#pragma unroll
        for (int t = 0; t < 8; ++t) {
            int d = t * 16 + l16;
            O[((size_t)(b * SEQ + s)) * DMODEL + h * DK + d] = f32_bf16(accO[t][r] * inv);
        }
    }
}

// ---------------------------------------------------------------- launch
extern "C" void kernel_launch(void* const* d_in, const int* in_sizes, int n_in,
                              void* d_out, int out_size, void* d_ws, size_t ws_size,
                              hipStream_t stream) {
    const float* x  = (const float*)d_in[0];
    const float* Wq = (const float*)d_in[1];
    const float* Wk = (const float*)d_in[2];
    const float* Wv = (const float*)d_in[3];
    const float* Wo = (const float*)d_in[4];
    float* out = (float*)d_out;

    const int BS = BATCH * SEQ;                 // 4096
    const size_t tile = (size_t)BS * DMODEL;    // 8,388,608 elems (bf16 -> 16.78 MB)
    unsigned short* xb = (unsigned short*)d_ws;
    unsigned short* Qb = xb + tile;
    unsigned short* Kb = Qb + tile;
    unsigned short* Vb = Kb + tile;
    unsigned short* Ob = Vb + tile;             // total ~84 MB of workspace

    {   // x -> bf16
        int n = (int)tile;
        convert_f32_bf16<<<n / (256 * 4), 256, 0, stream>>>(x, xb, n);
    }

    dim3 gg(DMODEL / 128, BS / 128);            // (16, 32)
    gemm_xwT<true><<<gg, 256, 0, stream>>>(xb, Wq, Qb, BS, DMODEL, DMODEL);
    gemm_xwT<true><<<gg, 256, 0, stream>>>(xb, Wk, Kb, BS, DMODEL, DMODEL);
    gemm_xwT<true><<<gg, 256, 0, stream>>>(xb, Wv, Vb, BS, DMODEL, DMODEL);

    {   // RoPE on Q and K
        int total = BS * (DMODEL / 2);
        int blocks = (total + 255) / 256;
        rope_inplace<<<blocks, 256, 0, stream>>>(Qb, BS);
        rope_inplace<<<blocks, 256, 0, stream>>>(Kb, BS);
    }

    {   // causal flash attention
        dim3 ga(SEQ / 64, BATCH * N_HEADS);     // (32, 32)
        flash_attn<<<ga, 128, 0, stream>>>(Qb, Kb, Vb, Ob);
    }

    // output projection -> f32 d_out
    gemm_xwT<false><<<gg, 256, 0, stream>>>(Ob, Wo, (void*)out, BS, DMODEL, DMODEL);
}
